// MultiMamba2Mixer_60507499266083
// MI455X (gfx1250) — matrix operand order repaired and checked by reading, not compiled, not run on hardware
//
#include <hip/hip_runtime.h>
#include <cstdint>

typedef __attribute__((ext_vector_type(16))) __bf16 v16bf;
typedef __attribute__((ext_vector_type(8)))  float  v8f;

#define DM 2048      // D_MODEL
#define HH 64        // heads
#define PP 64        // head dim
#define NN 128       // state dim
#define KK 4         // conv kernel
#define CHK 256      // chunk
#define DI 4096      // D_INNER
#define CD 4352      // CONV_DIM
#define NP 8512      // in_proj output width
#define BB 2
#define LL 2048
#define NT (BB*LL)   // 4096 tokens
#define NC (LL/CHK)  // 8 chunks per batch

// ---------------- helpers ----------------

__device__ __forceinline__ unsigned short f2bf(float f){
  unsigned int u = __float_as_uint(f);
  u += 0x7FFFu + ((u >> 16) & 1u);   // round-to-nearest-even
  return (unsigned short)(u >> 16);
}
__device__ __forceinline__ float bf2f(unsigned short h){
  return __uint_as_float(((unsigned int)h) << 16);
}
__device__ __forceinline__ v8f zerov8(){
  v8f z = {0.f,0.f,0.f,0.f,0.f,0.f,0.f,0.f};
  return z;
}
__device__ __forceinline__ v8f wmma_bf16(v16bf a, v16bf b, v8f c){
  return __builtin_amdgcn_wmma_f32_16x16x32_bf16(false, a, false, b, (short)0, c, false, false);
}
// A operand 16(M)x32(K): lane t -> row t&15, khalf t>>4.
// dword g: kbase = (g<4 ? 2g : 16+2(g-4)) + 8*khalf
__device__ __forceinline__ v16bf fragA(const unsigned short* s, int stride, int row, int lhalf){
  union { unsigned int u[8]; v16bf v; } f;
  const unsigned short* rp = s + row*stride;
#pragma unroll
  for (int g=0; g<8; ++g){
    int kb = ((g<4) ? (2*g) : (16 + 2*(g-4))) + 8*lhalf;
    f.u[g] = *(const unsigned int*)(rp + kb);
  }
  return f.v;
}
// B operand 32(K)x16(N): lane t -> col t&15, ksec t>>4.
// dword g: k = (g<4 ? 2g : 16+2(g-4)) + 8*ksec. LDS tile is [col][k] (K contiguous).
__device__ __forceinline__ v16bf fragB(const unsigned short* s, int stride, int col, int lhalf, int koff){
  union { unsigned int u[8]; v16bf v; } f;
  const unsigned short* cp = s + col*stride + koff;
#pragma unroll
  for (int g=0; g<8; ++g){
    f.u[g] = *(const unsigned int*)(cp + ((g<4) ? (2*g) : (16 + 2*(g-4))) + 8*lhalf);
  }
  return f.v;
}

// ---------------- generic bf16 WMMA GEMM ----------------
// C[M,Nd] = A[M,Kd] @ B  (BT=false: B is [Kd,Nd] row-major; BT=true: B is [Nd,Kd] row-major, used transposed)
// 128x128 block tile, 256 threads = 8 waves (4 M x 2 N), each wave 32x64 (2x4 wmma tiles).
// M, Kd multiples of 128/32; Nd multiple of 4 (guarded). float4 global loads, register prefetch of next K-slice.
template<bool BT, bool OUTBF>
__global__ void __launch_bounds__(256)
wmma_gemm(const float* __restrict__ A, const float* __restrict__ Bm,
          void* __restrict__ Cout, int M, int Nd, int Kd,
          int lda, int ldb, int ldc)
{
  __shared__ unsigned short As[128][34];
  __shared__ unsigned short Bt[128][34];
  const int tid  = threadIdx.x;
  const int lane = tid & 31, wave = tid >> 5;
  const int wr = wave >> 1, wc = wave & 1;      // 4 (M) x 2 (N)
  const int lrow = lane & 15, lhalf = lane >> 4;
  const int m0 = blockIdx.y * 128, n0 = blockIdx.x * 128;
  (void)M;

  v8f acc[2][4];
#pragma unroll
  for (int i=0;i<2;++i)
#pragma unroll
    for (int j=0;j<4;++j) acc[i][j] = zerov8();

  float4 ra[4], rb[4];

  auto loadA = [&](int k0){
#pragma unroll
    for (int q=0;q<4;++q){
      int idx = tid + q*256;
      int r = idx >> 3, c4 = idx & 7;
      ra[q] = *(const float4*)&A[(long)(m0 + r)*lda + k0 + c4*4];
    }
  };
  auto loadB = [&](int k0){
#pragma unroll
    for (int q=0;q<4;++q){
      int idx = tid + q*256;
      if (BT){
        int r = idx >> 3, c4 = idx & 7;
        rb[q] = (n0 + r < Nd) ? *(const float4*)&Bm[(long)(n0 + r)*ldb + k0 + c4*4]
                              : make_float4(0.f,0.f,0.f,0.f);
      } else {
        int k = idx >> 5, c4 = idx & 31;
        int col = n0 + c4*4;
        rb[q] = (col < Nd) ? *(const float4*)&Bm[(long)(k0 + k)*ldb + col]
                           : make_float4(0.f,0.f,0.f,0.f);
      }
    }
  };
  auto stage = [&](){
#pragma unroll
    for (int q=0;q<4;++q){
      int idx = tid + q*256;
      {
        int r = idx >> 3, c4 = idx & 7;
        unsigned int u0 = ((unsigned int)f2bf(ra[q].y) << 16) | (unsigned int)f2bf(ra[q].x);
        unsigned int u1 = ((unsigned int)f2bf(ra[q].w) << 16) | (unsigned int)f2bf(ra[q].z);
        *(unsigned int*)&As[r][c4*4]     = u0;
        *(unsigned int*)&As[r][c4*4 + 2] = u1;
      }
      if (BT){
        int r = idx >> 3, c4 = idx & 7;
        unsigned int u0 = ((unsigned int)f2bf(rb[q].y) << 16) | (unsigned int)f2bf(rb[q].x);
        unsigned int u1 = ((unsigned int)f2bf(rb[q].w) << 16) | (unsigned int)f2bf(rb[q].z);
        *(unsigned int*)&Bt[r][c4*4]     = u0;
        *(unsigned int*)&Bt[r][c4*4 + 2] = u1;
      } else {
        int k = idx >> 5, c4 = idx & 31;
        Bt[c4*4 + 0][k] = f2bf(rb[q].x);
        Bt[c4*4 + 1][k] = f2bf(rb[q].y);
        Bt[c4*4 + 2][k] = f2bf(rb[q].z);
        Bt[c4*4 + 3][k] = f2bf(rb[q].w);
      }
    }
  };

  loadA(0); loadB(0);
  for (int k0 = 0; k0 < Kd; k0 += 32){
    stage();
    __syncthreads();
    if (k0 + 32 < Kd){ loadA(k0 + 32); loadB(k0 + 32); }   // prefetch next K-slice into VGPRs
    v16bf a0 = fragA(&As[0][0], 34, wr*32 +      lrow, lhalf);
    v16bf a1 = fragA(&As[0][0], 34, wr*32 + 16 + lrow, lhalf);
#pragma unroll
    for (int nt = 0; nt < 4; ++nt){
      v16bf bfr = fragB(&Bt[0][0], 34, wc*64 + nt*16 + lrow, lhalf, 0);
      acc[0][nt] = wmma_bf16(a0, bfr, acc[0][nt]);
      acc[1][nt] = wmma_bf16(a1, bfr, acc[1][nt]);
    }
    __syncthreads();
  }
#pragma unroll
  for (int si=0; si<2; ++si)
#pragma unroll
  for (int nt=0; nt<4; ++nt)
#pragma unroll
  for (int r=0; r<8; ++r){
    int row = m0 + wr*32 + si*16 + r + 8*lhalf;
    int col = n0 + wc*64 + nt*16 + lrow;
    if (col < Nd){
      float v = acc[si][nt][r];
      if (OUTBF) ((unsigned short*)Cout)[(long)row*ldc + col] = f2bf(v);
      else       ((float*)Cout)[(long)row*ldc + col] = v;
    }
  }
}

// ---------------- depthwise causal conv (K=4) + bias + silu ----------------
__global__ void conv_silu_kernel(const float* __restrict__ proj,
                                 const float* __restrict__ cw,
                                 const float* __restrict__ cb,
                                 float* __restrict__ hbc)
{
  long idx = (long)blockIdx.x*blockDim.x + threadIdx.x;
  if (idx >= (long)NT*CD) return;
  int ch  = (int)(idx % CD);
  int tok = (int)(idx / CD);
  int l = tok % LL, b = tok / LL;
  float acc = cb[ch];
#pragma unroll
  for (int k=0;k<KK;++k){
    int ls = l - (KK-1) + k;
    if (ls >= 0) acc += cw[ch*KK + k] * proj[(long)(b*LL + ls)*NP + DI + ch];
  }
  hbc[(long)tok*CD + ch] = acc / (1.f + __expf(-acc));
}

// ---------------- softplus(dt + bias) ----------------
__global__ void dt_kernel(const float* __restrict__ proj,
                          const float* __restrict__ dt_bias,
                          float* __restrict__ dts)
{
  int idx = blockIdx.x*blockDim.x + threadIdx.x;
  if (idx >= NT*HH) return;
  int h = idx & (HH-1); int tok = idx >> 6;
  float v = proj[(long)tok*NP + DI + CD + h] + dt_bias[h];
  dts[idx] = (v > 20.f) ? v : log1pf(__expf(v));
}

// ---------------- per-(b,c,h) cumsum of dA, chunk decay ----------------
__global__ void cum_kernel(const float* __restrict__ dts,
                           const float* __restrict__ A_log,
                           float* __restrict__ cum, float* __restrict__ cdec)
{
  int idx = blockIdx.x*blockDim.x + threadIdx.x;
  if (idx >= BB*NC*HH) return;
  int h = idx % HH; int bc = idx / HH; int c = bc % NC; int b = bc / NC;
  float A = -__expf(A_log[h]);
  float run = 0.f;
  int tok0 = b*LL + c*CHK;
  for (int i=0;i<CHK;++i){
    run += dts[(tok0+i)*HH + h]*A;
    cum[(long)idx*CHK + i] = run;
  }
  cdec[idx] = __expf(run);
}

// ---------------- per-(b,c,h) chunk state: states[n][p] = sum_l B[l][n]*x[l][p]*dt_l*exp(cumLast-cum_l)
__global__ void __launch_bounds__(256)
states_kernel(const float* __restrict__ hbc, const float* __restrict__ dts,
              const float* __restrict__ cum, float* __restrict__ states)
{
  __shared__ unsigned short As[128][34];
  __shared__ unsigned short Xt[64][34];
  __shared__ float scale[CHK];
  int bz = blockIdx.x;
  int h = bz % HH; int bc = bz / HH; int c = bc % NC; int b = bc / NC;
  int tid = threadIdx.x;
  int lane = tid & 31, wave = tid >> 5;
  int wr = wave >> 1, wc = wave & 1;          // 4 x 2 waves -> 128 x 64
  int lrow = lane & 15, lhalf = lane >> 4;
  int tok0 = b*LL + c*CHK;
  long cumbase = (long)bz*CHK;

  float cumLast = cum[cumbase + CHK - 1];
  for (int i = tid; i < CHK; i += 256){
    scale[i] = dts[(tok0 + i)*HH + h] * __expf(cumLast - cum[cumbase + i]);
  }
  v8f acc[2][2];
#pragma unroll
  for (int i=0;i<2;++i)
#pragma unroll
    for (int j=0;j<2;++j) acc[i][j] = zerov8();

  for (int l0 = 0; l0 < CHK; l0 += 32){
    __syncthreads();
    for (int idx = tid; idx < 128*32; idx += 256){
      int n = idx & 127, ll = idx >> 7;
      As[n][ll] = f2bf(hbc[(long)(tok0 + l0 + ll)*CD + DI + n]);      // B^T
    }
    for (int idx = tid; idx < 64*32; idx += 256){
      int p = idx & 63, ll = idx >> 6;
      float x = hbc[(long)(tok0 + l0 + ll)*CD + h*PP + p];
      Xt[p][ll] = f2bf(x * scale[l0 + ll]);
    }
    __syncthreads();
    v16bf a0 = fragA(&As[0][0], 34, wr*32 +      lrow, lhalf);
    v16bf a1 = fragA(&As[0][0], 34, wr*32 + 16 + lrow, lhalf);
    v16bf b0 = fragB(&Xt[0][0], 34, wc*32 +      lrow, lhalf, 0);
    v16bf b1 = fragB(&Xt[0][0], 34, wc*32 + 16 + lrow, lhalf, 0);
    acc[0][0] = wmma_bf16(a0, b0, acc[0][0]);
    acc[0][1] = wmma_bf16(a0, b1, acc[0][1]);
    acc[1][0] = wmma_bf16(a1, b0, acc[1][0]);
    acc[1][1] = wmma_bf16(a1, b1, acc[1][1]);
  }
  long sbase = (long)bz * (PP*NN);
#pragma unroll
  for (int si=0; si<2; ++si)
#pragma unroll
  for (int sj=0; sj<2; ++sj)
#pragma unroll
  for (int r=0; r<8; ++r){
    int n = wr*32 + si*16 + r + 8*lhalf;
    int p = wc*32 + sj*16 + lrow;
    states[sbase + (long)n*PP + p] = acc[si][sj][r];
  }
}

// ---------------- sequential inter-chunk recurrence (8 chunks) ----------------
__global__ void recurrence_kernel(const float* __restrict__ states,
                                  const float* __restrict__ cdec,
                                  float* __restrict__ prevs)
{
  int bh = blockIdx.x;
  int b = bh / HH, h = bh % HH;
  int tid = threadIdx.x;
  for (int e = tid; e < PP*NN; e += 256){
    float pv = 0.f;
    for (int c = 0; c < NC; ++c){
      long bz = (long)(b*NC + c)*HH + h;
      long o = bz*(PP*NN) + e;
      prevs[o] = pv;
      pv = cdec[bz]*pv + states[o];
    }
  }
}

// ---------------- per-(b,c,h): Y = (Sraw * decay * dt) @ X + (C*exp(cum)) @ prev^T + D*x
__global__ void __launch_bounds__(256)
yh_kernel(const float* __restrict__ hbc, const float* __restrict__ dts,
          const float* __restrict__ cum, const unsigned short* __restrict__ Sraw,
          const float* __restrict__ prevs, const float* __restrict__ Dp,
          float* __restrict__ yh)
{
  __shared__ unsigned short Xt[64][258];    // x^T bf16, full chunk [p][l]
  __shared__ unsigned short As[256][34];    // A-operand staging
  __shared__ unsigned short Pt[64][34];     // prev^T staging [p][n]
  __shared__ float cumS[CHK];
  __shared__ float dtS[CHK];
  int bz = blockIdx.x;
  int h = bz % HH; int bc = bz / HH; int c = bc % NC; int b = bc / NC;
  int tid = threadIdx.x;
  int lane = tid & 31, wave = tid >> 5;     // 8 waves, each 32 rows x 64 cols
  int lrow = lane & 15, lhalf = lane >> 4;
  int tok0 = b*LL + c*CHK;
  long cumbase = (long)bz*CHK;

  for (int i = tid; i < CHK; i += 256){
    cumS[i] = cum[cumbase + i];
    dtS[i]  = dts[(tok0 + i)*HH + h];
  }
  for (int idx = tid; idx < CHK*PP; idx += 256){
    int j = idx >> 6, p = idx & 63;
    Xt[p][j] = f2bf(hbc[(long)(tok0 + j)*CD + h*PP + p]);
  }
  v8f acc[2][4];
#pragma unroll
  for (int i=0;i<2;++i)
#pragma unroll
    for (int j=0;j<4;++j) acc[i][j] = zerov8();

  const unsigned short* Sb = Sraw + (long)bc * (CHK*CHK);

  // intra-chunk (diag) term: flash-style over j tiles of 32
  for (int j0 = 0; j0 < CHK; j0 += 32){
    __syncthreads();
    for (int idx = tid; idx < CHK*32; idx += 256){
      int i = idx >> 5, jj = idx & 31;
      int j = j0 + jj;
      float v = 0.f;
      if (i >= j){
        float s = bf2f(Sb[(long)i*CHK + j]);
        v = s * dtS[j] * __expf(cumS[i] - cumS[j]);
      }
      As[i][jj] = f2bf(v);
    }
    __syncthreads();
    if (j0 <= wave*32 + 31){      // wave-uniform skip of all-zero tiles
      v16bf a0 = fragA(&As[0][0], 34, wave*32 +      lrow, lhalf);
      v16bf a1 = fragA(&As[0][0], 34, wave*32 + 16 + lrow, lhalf);
#pragma unroll
      for (int pj = 0; pj < 4; ++pj){
        v16bf bfr = fragB(&Xt[0][0], 258, pj*16 + lrow, lhalf, j0);
        acc[0][pj] = wmma_bf16(a0, bfr, acc[0][pj]);
        acc[1][pj] = wmma_bf16(a1, bfr, acc[1][pj]);
      }
    }
  }

  // inter-chunk (off) term: C[i][n]*exp(cum_i) @ prev[p][n]^T over n tiles of 32
  long pbase = (long)bz * (PP*NN);
  for (int n0 = 0; n0 < NN; n0 += 32){
    __syncthreads();
    for (int idx = tid; idx < CHK*32; idx += 256){
      int i = idx >> 5, nn = idx & 31;
      float cv = hbc[(long)(tok0 + i)*CD + DI + NN + (n0 + nn)];
      As[i][nn] = f2bf(cv * __expf(cumS[i]));
    }
    for (int idx = tid; idx < PP*32; idx += 256){
      int nn = idx >> 6, p = idx & 63;
      Pt[p][nn] = f2bf(prevs[pbase + (long)(n0 + nn)*PP + p]);
    }
    __syncthreads();
    v16bf a0 = fragA(&As[0][0], 34, wave*32 +      lrow, lhalf);
    v16bf a1 = fragA(&As[0][0], 34, wave*32 + 16 + lrow, lhalf);
#pragma unroll
    for (int pj = 0; pj < 4; ++pj){
      v16bf bfr = fragB(&Pt[0][0], 34, pj*16 + lrow, lhalf, 0);
      acc[0][pj] = wmma_bf16(a0, bfr, acc[0][pj]);
      acc[1][pj] = wmma_bf16(a1, bfr, acc[1][pj]);
    }
  }
  __syncthreads();

  float Dv = Dp[h];
#pragma unroll
  for (int si=0; si<2; ++si)
#pragma unroll
  for (int pj=0; pj<4; ++pj)
#pragma unroll
  for (int r=0; r<8; ++r){
    int i = wave*32 + si*16 + r + 8*lhalf;
    int p = pj*16 + lrow;
    float xv = bf2f(Xt[p][i]);
    yh[(long)(tok0 + i)*DI + h*PP + p] = acc[si][pj][r] + Dv*xv;
  }
}

// ---------------- gated silu + RMS norm (in-place on yh) ----------------
__global__ void __launch_bounds__(256)
norm_kernel(float* __restrict__ yh, const float* __restrict__ proj,
            const float* __restrict__ nw)
{
  __shared__ float hrow[DI];
  __shared__ float red[256];
  int tok = blockIdx.x, tid = threadIdx.x;
  float ss = 0.f;
  for (int d = tid; d < DI; d += 256){
    float g  = proj[(long)tok*NP + d];
    float hv = yh[(long)tok*DI + d] * (g / (1.f + __expf(-g)));
    hrow[d] = hv; ss += hv*hv;
  }
  red[tid] = ss; __syncthreads();
  for (int s = 128; s > 0; s >>= 1){
    if (tid < s) red[tid] += red[tid + s];
    __syncthreads();
  }
  float scl = rsqrtf(red[0]/(float)DI + 1e-5f);
  for (int d = tid; d < DI; d += 256){
    yh[(long)tok*DI + d] = hrow[d]*scl*nw[d];
  }
}

// ---------------- host launcher ----------------
extern "C" void kernel_launch(void* const* d_in, const int* in_sizes, int n_in,
                              void* d_out, int out_size, void* d_ws, size_t ws_size,
                              hipStream_t stream)
{
  (void)in_sizes; (void)n_in; (void)out_size; (void)ws_size;
  const float* hidden  = (const float*)d_in[0];
  const float* in_w    = (const float*)d_in[1];
  const float* conv_w  = (const float*)d_in[2];
  const float* conv_b  = (const float*)d_in[3];
  const float* dt_bias = (const float*)d_in[4];
  const float* A_log   = (const float*)d_in[5];
  const float* Dp      = (const float*)d_in[6];
  const float* norm_w  = (const float*)d_in[7];
  const float* out_w   = (const float*)d_in[8];
  float* out = (float*)d_out;

  char* ws = (char*)d_ws;
  size_t off = 0;
  auto alloc = [&](size_t bytes)->char* {
    char* p = ws + off;
    off += (bytes + 255) & ~(size_t)255;
    return p;
  };
  float* proj   = (float*)alloc((size_t)NT*NP*4);              // 139.5 MB
  float* hbc    = (float*)alloc((size_t)NT*CD*4);              // 71.3 MB
  float* dts    = (float*)alloc((size_t)NT*HH*4);              // 1 MB
  float* cum    = (float*)alloc((size_t)BB*NC*HH*CHK*4);       // 1 MB
  float* cdec   = (float*)alloc((size_t)BB*NC*HH*4);
  unsigned short* Sraw = (unsigned short*)alloc((size_t)BB*NC*CHK*CHK*2);  // 2 MB
  float* states = (float*)alloc((size_t)BB*NC*HH*PP*NN*4);     // 33.5 MB
  float* prevs  = (float*)alloc((size_t)BB*NC*HH*PP*NN*4);     // 33.5 MB
  float* yh     = (float*)alloc((size_t)NT*DI*4);              // 67 MB

  // 1. in_proj GEMM: proj[NT,NP] = hidden[NT,DM] @ in_w[DM,NP]
  {
    dim3 g((NP + 127)/128, NT/128);   // 67 x 32
    wmma_gemm<false,false><<<g, 256, 0, stream>>>(hidden, in_w, proj, NT, NP, DM, DM, NP, NP);
  }
  // 2. depthwise causal conv + bias + silu
  {
    long n = (long)NT*CD;
    conv_silu_kernel<<<(unsigned)((n + 255)/256), 256, 0, stream>>>(proj, conv_w, conv_b, hbc);
  }
  // 3. softplus(dt + bias)
  dt_kernel<<<(NT*HH + 255)/256, 256, 0, stream>>>(proj, dt_bias, dts);
  // 4. per-chunk cumsum of dA + chunk decay
  cum_kernel<<<(BB*NC*HH + 127)/128, 128, 0, stream>>>(dts, A_log, cum, cdec);
  // 5. Sraw[b,c] = C @ B^T  (shared across all 64 heads since G=1)
  for (int b = 0; b < BB; ++b)
    for (int c = 0; c < NC; ++c){
      const float* Cp = hbc + (size_t)(b*LL + c*CHK)*CD + DI + NN;
      const float* Bp = hbc + (size_t)(b*LL + c*CHK)*CD + DI;
      unsigned short* Sp = Sraw + (size_t)(b*NC + c)*CHK*CHK;
      dim3 g(CHK/128, CHK/128);       // 2 x 2
      wmma_gemm<true,true><<<g, 256, 0, stream>>>(Cp, Bp, Sp, CHK, CHK, NN, CD, CD, CHK);
    }
  // 6. per-chunk states
  states_kernel<<<BB*NC*HH, 256, 0, stream>>>(hbc, dts, cum, states);
  // 7. inter-chunk recurrence
  recurrence_kernel<<<BB*HH, 256, 0, stream>>>(states, cdec, prevs);
  // 8. Y = diag + off + D*x
  yh_kernel<<<BB*NC*HH, 256, 0, stream>>>(hbc, dts, cum, Sraw, prevs, Dp, yh);
  // 9. gated RMS norm (in-place)
  norm_kernel<<<NT, 256, 0, stream>>>(yh, proj, norm_w);
  // 10. out_proj GEMM: out[NT,DM] = yh[NT,DI] @ out_w[DI,DM]
  {
    dim3 g(DM/128, NT/128);           // 16 x 32
    wmma_gemm<false,false><<<g, 256, 0, stream>>>(yh, out_w, out, NT, DM, DI, DI, DM, DM);
  }
}
